// RotarySelfAttention_8134668058983
// MI455X (gfx1250) — matrix operand order repaired and checked by reading, not compiled, not run on hardware
//
#include <hip/hip_runtime.h>

typedef __bf16 v16bf __attribute__((ext_vector_type(16)));
typedef float  v8f   __attribute__((ext_vector_type(8)));

union F16 { v16bf v; unsigned u[8]; };                  // 16 bf16 <-> 8 dwords
union PK2 { __bf16 h[2]; unsigned u; };
union PK4 { __bf16 h[4]; unsigned long long u; };
union CVT { __bf16 h; unsigned short s; };

#if __has_builtin(__builtin_amdgcn_sched_barrier)
#define SCHED_FENCE() __builtin_amdgcn_sched_barrier(0)
#else
#define SCHED_FENCE()
#endif

__device__ inline void st_out(float* p, float v)  { *p = v; }
__device__ inline void st_out(__bf16* p, float v) { *p = (__bf16)v; }

// ---------------------------------------------------------------------------
// Pack f32 -> bf16 row-major (vectorized: float4 in, u64 out). n4 = elems/4.
// ---------------------------------------------------------------------------
__global__ __launch_bounds__(256)
void pack_bf16(const float* __restrict__ src, __bf16* __restrict__ dst, int n4)
{
    int g = blockIdx.x * blockDim.x + threadIdx.x;
    if (g >= n4) return;
    float4 v = ((const float4*)src)[g];
    PK4 pk;
    pk.h[0] = (__bf16)v.x; pk.h[1] = (__bf16)v.y;
    pk.h[2] = (__bf16)v.z; pk.h[3] = (__bf16)v.w;
    ((unsigned long long*)dst)[g] = pk.u;
}

// ---------------------------------------------------------------------------
// Pack weight W[K][Nc] (f32, row-major) into B-fragment layout:
// word[((nc*(K/32) + kc)*32 + lane)*8 + j] = ( W[k][col], W[k+1][col] ) bf16
//   where lane = half*16 + bn, k = kc*32 + half*16 + 2j, col = nc*16 + bn.
// ---------------------------------------------------------------------------
__global__ __launch_bounds__(256)
void pack_w(const float* __restrict__ W, unsigned* __restrict__ Wp, int K, int Nc)
{
    int idx = blockIdx.x * blockDim.x + threadIdx.x;
    int total = (K / 2) * Nc;
    if (idx >= total) return;
    int j    = idx & 7;
    int lane = (idx >> 3) & 31;
    int rest = idx >> 8;
    int kc   = rest % (K / 32);
    int nc   = rest / (K / 32);
    int half = lane >> 4, bn = lane & 15;
    int k    = kc * 32 + half * 16 + 2 * j;
    int col  = nc * 16 + bn;
    PK2 pk;
    pk.h[0] = (__bf16)W[(size_t)k * Nc + col];
    pk.h[1] = (__bf16)W[(size_t)(k + 1) * Nc + col];
    Wp[idx] = pk.u;
}

// ---------------------------------------------------------------------------
// Register-resident GEMM, software-pipelined; sched barriers pin the
// next-step fragment loads ahead of the current-step WMMAs so loads stay
// in flight during matrix execution.
// ---------------------------------------------------------------------------
template <typename OT>
__global__ __launch_bounds__(256)
void gemm_wmma(const __bf16* __restrict__ A, const unsigned* __restrict__ Wp,
               OT* __restrict__ C, int M, int K, int Nc)
{
    const int tid  = threadIdx.x;
    const int wave = tid >> 5;
    const int lane = tid & 31;
    const int m0 = blockIdx.y * 128;
    const int n0 = blockIdx.x * 64;

    const int am = lane & 15;
    const int ak = (lane < 16) ? 0 : 16;
    const __bf16* Arow = A + (size_t)(m0 + wave * 16 + am) * K;
    const int KC = K >> 5;

    v8f acc[4];
    #pragma unroll
    for (int c = 0; c < 4; ++c)
        #pragma unroll
        for (int i = 0; i < 8; ++i) acc[c][i] = 0.0f;

    auto load_a = [&](F16& a, int kc) {
        const __bf16* ap = Arow + kc * 32 + ak;
        #pragma unroll
        for (int j = 0; j < 4; ++j) {
            a.u[j]     = *(const unsigned*)&ap[2 * j];
            a.u[4 + j] = *(const unsigned*)&ap[8 + 2 * j];
        }
    };
    auto load_b = [&](F16* bf, int kc) {
        #pragma unroll
        for (int c = 0; c < 4; ++c) {
            const unsigned* bp =
                Wp + (((size_t)(n0 / 16 + c) * KC + kc) * 32 + lane) * 8;
            #pragma unroll
            for (int j = 0; j < 8; ++j) bf[c].u[j] = bp[j];
        }
    };
    auto mm4 = [&](F16& a, F16* bf) {
        #pragma unroll
        for (int c = 0; c < 4; ++c)
            acc[c] = __builtin_amdgcn_wmma_f32_16x16x32_bf16(
                false, a.v, false, bf[c].v, (short)0, acc[c], false, false);
    };

    F16 aA, aB, bA[4], bB[4];
    load_a(aA, 0);
    load_b(bA, 0);
    for (int kc = 0; kc < KC; kc += 2) {
        if (kc + 4 < KC)
            __builtin_prefetch(Arow + (kc + 4) * 32, 0, 0);
        if (kc + 1 < KC) { load_a(aB, kc + 1); load_b(bB, kc + 1); }
        SCHED_FENCE();           // loads for kc+1 issue before WMMAs for kc
        mm4(aA, bA);
        if (kc + 2 < KC) { load_a(aA, kc + 2); load_b(bA, kc + 2); }
        SCHED_FENCE();           // loads for kc+2 issue before WMMAs for kc+1
        if (kc + 1 < KC) mm4(aB, bB);
    }

    const int mh = (lane < 16) ? 0 : 8;
    #pragma unroll
    for (int c = 0; c < 4; ++c)
        #pragma unroll
        for (int r = 0; r < 8; ++r)
            st_out(&C[(size_t)(m0 + wave * 16 + mh + r) * Nc + n0 + c * 16 + am],
                   acc[c][r]);
}

// ---------------------------------------------------------------------------
// RoPE + head split. qkv bf16 [B,N,3,H,hd]. Q,K rotated -> [B,H,N,hd] (u32
// pair stores). V -> B-fragment-packed layout Vp[bh][N/32][4][lane][8].
// ---------------------------------------------------------------------------
__global__ __launch_bounds__(256)
void rope_split(const __bf16* __restrict__ qkv,
                __bf16* __restrict__ Q, __bf16* __restrict__ Kt,
                unsigned short* __restrict__ Vp, int B, int N, int H, int hd)
{
    int idx = blockIdx.x * blockDim.x + threadIdx.x;
    int hp = hd / 2;
    int total = B * N * H * hp;
    if (idx >= total) return;
    int i = idx % hp;  int t = idx / hp;
    int h = t % H;     t /= H;
    int n = t % N;     int b = t / N;
    int D = H * hd;

    const float LOG1E4 = 9.210340371976184f;  // ln(10000)
    float freq = __expf(-((float)(2 * i) / (float)hd) * LOG1E4);
    float ang = (float)n * freq;
    float s = __sinf(ang), c = __cosf(ang);

    size_t src = ((size_t)(b * N + n)) * 3 * D + h * hd + 2 * i;
    float q0 = (float)qkv[src],         q1 = (float)qkv[src + 1];
    float k0 = (float)qkv[src + D],     k1 = (float)qkv[src + D + 1];
    float v0 = (float)qkv[src + 2 * D], v1 = (float)qkv[src + 2 * D + 1];

    size_t dst = (((size_t)(b * H + h)) * N + n) * hd + 2 * i;
    PK2 pq; pq.h[0] = (__bf16)(q0 * c - q1 * s); pq.h[1] = (__bf16)(q1 * c + q0 * s);
    ((unsigned*)Q)[dst >> 1] = pq.u;
    PK2 pk; pk.h[0] = (__bf16)(k0 * c - k1 * s); pk.h[1] = (__bf16)(k1 * c + k0 * s);
    ((unsigned*)Kt)[dst >> 1] = pk.u;

    int kb32 = n >> 5, key_in = n & 31;
    int half = key_in >> 4, tt = key_in & 15;
    int jj = tt >> 1, par = tt & 1;
    float vv[2] = { v0, v1 };
    #pragma unroll
    for (int e = 0; e < 2; ++e) {
        int d = 2 * i + e;
        int cc = d >> 4, bn = d & 15;
        int vlane = half * 16 + bn;
        size_t word =
            ((((size_t)(b * H + h) * (N >> 5) + kb32) * 4 + cc) * 32 + vlane) * 8 + jj;
        CVT cv; cv.h = (__bf16)vv[e];
        Vp[word * 2 + par] = cv.s;
    }
}

// ---------------------------------------------------------------------------
// Flash attention: block = (64 q rows, h, b), 4 waves; wave owns 16 q rows.
// 64-key blocks. Sched barriers pin: (1) all 16 K-frag loads before the 8
// QK^T WMMAs (staged waits), (2) all V-frag loads before the softmax so
// their latency hides under it; PV WMMAs then run from registers.
// ---------------------------------------------------------------------------
__global__ __launch_bounds__(128)
void attn_flash(const __bf16* __restrict__ Q, const __bf16* __restrict__ Kt,
                const unsigned* __restrict__ Vp, __bf16* __restrict__ Ao,
                int B, int N, int H, int hd)
{
    const int h = blockIdx.y, b = blockIdx.z;
    const int wave = threadIdx.x >> 5, lane = threadIdx.x & 31;
    const int q0 = blockIdx.x * 64 + wave * 16;

    const size_t bh = (size_t)(b * H + h);
    const __bf16* Qb = Q + bh * N * hd;
    const __bf16* Kb = Kt + bh * N * hd;
    const unsigned* Vpb = Vp + bh * (size_t)(N >> 5) * 4 * 32 * 8;

    __shared__ __bf16 p_lds[4][16][72];   // padded: 144B row stride

    const int am = lane & 15;
    const int ak = (lane < 16) ? 0 : 16;
    const int bk = (lane < 16) ? 0 : 16;
    const int mh = (lane < 16) ? 0 : 8;
    const float scale = 0.125f;  // hd^-0.5

    // Q fragments for the whole key loop: two 16x32 A-frags (hd = 64)
    F16 qf[2];
    const __bf16* qrow = Qb + (size_t)(q0 + am) * hd;
    #pragma unroll
    for (int c2 = 0; c2 < 2; ++c2) {
        const __bf16* qp = qrow + c2 * 32 + ak;
        #pragma unroll
        for (int j = 0; j < 4; ++j) {
            qf[c2].u[j]     = *(const unsigned*)&qp[2 * j];
            qf[c2].u[4 + j] = *(const unsigned*)&qp[8 + 2 * j];
        }
    }

    float mrow[8], lrow[8];
    v8f o[4];
    #pragma unroll
    for (int r = 0; r < 8; ++r) { mrow[r] = -1e30f; lrow[r] = 0.0f; }
    #pragma unroll
    for (int c = 0; c < 4; ++c)
        #pragma unroll
        for (int i = 0; i < 8; ++i) o[c][i] = 0.0f;

    for (int kb = 0; kb < N; kb += 64) {
        if (kb + 64 < N)
            __builtin_prefetch(Kb + (size_t)(kb + 64 + am) * hd, 0, 0);

        // (1) Preload all eight K fragments; barrier keeps them ahead of the
        //     WMMAs so the 16 b128 loads clause and pipeline.
        F16 kf[4][2];
        #pragma unroll
        for (int s2 = 0; s2 < 4; ++s2) {
            const __bf16* krow = Kb + (size_t)(kb + s2 * 16 + am) * hd;
            #pragma unroll
            for (int c2 = 0; c2 < 2; ++c2) {
                const __bf16* kp = krow + c2 * 32 + bk;
                #pragma unroll
                for (int j = 0; j < 8; ++j)
                    kf[s2][c2].u[j] = *(const unsigned*)&kp[2 * j];
            }
        }
        SCHED_FENCE();

        // (2) S = Q*K^T: 8 WMMAs with staged waits
        v8f s[4];
        #pragma unroll
        for (int s2 = 0; s2 < 4; ++s2) {
            #pragma unroll
            for (int i = 0; i < 8; ++i) s[s2][i] = 0.0f;
            #pragma unroll
            for (int c2 = 0; c2 < 2; ++c2)
                s[s2] = __builtin_amdgcn_wmma_f32_16x16x32_bf16(
                    false, qf[c2].v, false, kf[s2][c2].v, (short)0, s[s2],
                    false, false);
        }

        // (3) Kick off all V-fragment loads; barrier stops them sinking past
        //     the softmax, so their latency hides under it.
        F16 vf[2][4];
        #pragma unroll
        for (int h64 = 0; h64 < 2; ++h64) {
            const unsigned* vbase =
                Vpb + ((size_t)((kb >> 5) + h64) * 4) * 256 + (size_t)lane * 8;
            #pragma unroll
            for (int c = 0; c < 4; ++c) {
                const unsigned* vp = vbase + (size_t)c * 256;
                #pragma unroll
                for (int j = 0; j < 8; ++j) vf[h64][c].u[j] = vp[j];
            }
        }
        SCHED_FENCE();

        // (4) Online softmax across 64 columns
        float alpha[8];
        #pragma unroll
        for (int r = 0; r < 8; ++r) {
            float v = fmaxf(fmaxf(s[0][r], s[1][r]), fmaxf(s[2][r], s[3][r])) * scale;
            #pragma unroll
            for (int msk = 8; msk >= 1; msk >>= 1)
                v = fmaxf(v, __shfl_xor(v, msk, 32));
            float mnew = fmaxf(mrow[r], v);
            alpha[r] = __expf(mrow[r] - mnew);
            mrow[r] = mnew;
            float rs = 0.0f;
            #pragma unroll
            for (int s2 = 0; s2 < 4; ++s2) {
                float p = __expf(s[s2][r] * scale - mnew);
                s[s2][r] = p;
                rs += p;
            }
            #pragma unroll
            for (int msk = 8; msk >= 1; msk >>= 1)
                rs += __shfl_xor(rs, msk, 32);
            lrow[r] = lrow[r] * alpha[r] + rs;
            #pragma unroll
            for (int c = 0; c < 4; ++c) o[c][r] *= alpha[r];
        }

        // (5) P: C-layout -> LDS (wave-private, padded) -> two A-frags
        #pragma unroll
        for (int r = 0; r < 8; ++r)
            #pragma unroll
            for (int s2 = 0; s2 < 4; ++s2)
                p_lds[wave][mh + r][s2 * 16 + am] = (__bf16)s[s2][r];

        // (6) O += P * V from registers (only the LDS pf dependency remains)
        #pragma unroll
        for (int h64 = 0; h64 < 2; ++h64) {
            F16 pf;
            const __bf16* pp = &p_lds[wave][am][h64 * 32 + ak];
            #pragma unroll
            for (int j = 0; j < 4; ++j) {
                pf.u[j]     = *(const unsigned*)&pp[2 * j];
                pf.u[4 + j] = *(const unsigned*)&pp[8 + 2 * j];
            }
            #pragma unroll
            for (int c = 0; c < 4; ++c)
                o[c] = __builtin_amdgcn_wmma_f32_16x16x32_bf16(
                    false, pf.v, false, vf[h64][c].v, (short)0, o[c],
                    false, false);
        }
    }

    // Normalize (reciprocal once per row), store to attn_out [B, N, D]
    float linv[8];
    #pragma unroll
    for (int r = 0; r < 8; ++r) linv[r] = 1.0f / lrow[r];
    const int D = H * hd;
    #pragma unroll
    for (int c = 0; c < 4; ++c)
        #pragma unroll
        for (int r = 0; r < 8; ++r)
            Ao[((size_t)b * N + q0 + mh + r) * D + h * hd + c * 16 + am] =
                (__bf16)(o[c][r] * linv[r]);
}

// ---------------------------------------------------------------------------
extern "C" void kernel_launch(void* const* d_in, const int* in_sizes, int n_in,
                              void* d_out, int out_size, void* d_ws, size_t ws_size,
                              hipStream_t stream)
{
    const float* x      = (const float*)d_in[0];
    const float* w_qkv  = (const float*)d_in[1];
    const float* w_proj = (const float*)d_in[2];
    float* out = (float*)d_out;

    const int B = 2, N = 2048, D = 1024, H = 16, hd = 64;
    const int M = B * N;  // 4096

    char* ws = (char*)d_ws;
    __bf16*   xbf   = (__bf16*)ws;                 ws += (size_t)M * D * 2;
    unsigned* WqkvP = (unsigned*)ws;               ws += (size_t)(D / 2) * 3 * D * 4;
    unsigned* WprjP = (unsigned*)ws;               ws += (size_t)(D / 2) * D * 4;
    __bf16*   qkv   = (__bf16*)ws;                 ws += (size_t)M * 3 * D * 2;
    __bf16*   Qb    = (__bf16*)ws;                 ws += (size_t)M * D * 2;
    __bf16*   Kb    = (__bf16*)ws;                 ws += (size_t)M * D * 2;
    unsigned* Vp    = (unsigned*)ws;               ws += (size_t)M * D * 2;
    __bf16*   Ao    = (__bf16*)ws;

    // 0) operand packing
    int n4 = M * D / 4;
    pack_bf16<<<(n4 + 255) / 256, 256, 0, stream>>>(x, xbf, n4);
    int wq = (D / 2) * 3 * D;
    pack_w<<<(wq + 255) / 256, 256, 0, stream>>>(w_qkv, WqkvP, D, 3 * D);
    int wp = (D / 2) * D;
    pack_w<<<(wp + 255) / 256, 256, 0, stream>>>(w_proj, WprjP, D, D);

    // 1) qkv = x @ w_qkv  (bf16 out)
    gemm_wmma<__bf16><<<dim3(3 * D / 64, M / 128), dim3(256), 0, stream>>>(
        xbf, WqkvP, qkv, M, D, 3 * D);

    // 2) RoPE + split into Q,K [B,H,N,hd] and packed Vp
    int pairs = B * N * H * (hd / 2);
    rope_split<<<(pairs + 255) / 256, dim3(256), 0, stream>>>(
        qkv, Qb, Kb, (unsigned short*)Vp, B, N, H, hd);

    // 3) flash attention -> Ao [B,N,D] bf16
    attn_flash<<<dim3(N / 64, H, B), dim3(128), 0, stream>>>(
        Qb, Kb, Vp, Ao, B, N, H, hd);

    // 4) out = Ao @ w_proj  (f32 out)
    gemm_wmma<float><<<dim3(D / 64, M / 128), dim3(256), 0, stream>>>(
        Ao, WprjP, out, M, D, D);
}
